// MultiHeadAttention_75093208203662
// MI455X (gfx1250) — matrix operand order repaired and checked by reading, not compiled, not run on hardware
//
#include <hip/hip_runtime.h>
#include <hip/hip_bf16.h>
#include <math.h>

typedef _Float16 half_t;
typedef __attribute__((ext_vector_type(16))) _Float16 v16h;
typedef __attribute__((ext_vector_type(8)))  _Float16 v8h;
typedef __attribute__((ext_vector_type(8)))  float    v8f;

#define B_  2
#define S_  2048
#define D_  1024
#define H_  16
#define DK_ 64
#define DV_ 64
#define BS_ (B_*S_)   // 4096
#define HD_ (H_*DV_)  // 1024

union V16u { v16h v; v8h h[2]; };

__device__ __forceinline__ v8f vzero() {
  v8f z;
#pragma unroll
  for (int i = 0; i < 8; ++i) z[i] = 0.0f;
  return z;
}

// A operand: 16x32 f16 tile from row-major src (stride ld), per CDNA5 A-layout:
// lanes 0-15 (row m=lane): a[0..7]=src[m][c0..c0+7],   a[8..15]=src[m][c0+16..23]
// lanes 16-31 (row m)    : a[0..7]=src[m][c0+8..15],   a[8..15]=src[m][c0+24..31]
__device__ __forceinline__ v16h load_A16x32(const half_t* base, int ld, int row0, int col0, int lane) {
  const int hi = lane >> 4, m = lane & 15;
  const half_t* p = base + (size_t)(row0 + m) * ld + col0 + 8 * hi;
  V16u u;
  u.h[0] = *(const v8h*)(p);
  u.h[1] = *(const v8h*)(p + 16);
  return u.v;
}

// B operand: 32x16 (K x N) tile where B is stored K-major/transposed:
// baseT[n][k] row-major with stride ld.  lane 16*hi+n holds b[i] = B[K=8*hi+i][n] (i<8), b[i] = B[K=16+8*hi+(i-8)][n] (i>=8).
__device__ __forceinline__ v16h load_B32x16(const half_t* baseT, int ld, int n0, int k0, int lane) {
  const int hi = lane >> 4, n = lane & 15;
  const half_t* p = baseT + (size_t)(n0 + n) * ld + k0 + 8 * hi;
  V16u u;
  u.h[0] = *(const v8h*)(p);
  u.h[1] = *(const v8h*)(p + 16);
  return u.v;
}

__device__ __forceinline__ v8f wmma16(v16h a, v16h b, v8f c) {
  return __builtin_amdgcn_wmma_f32_16x16x32_f16(false, a, false, b, (short)0, c, false, false);
}

// ---------------- prep kernels ----------------

__global__ void k_cvt_f16(const float* __restrict__ src, half_t* __restrict__ dst, int n) {
  int i = blockIdx.x * blockDim.x + threadIdx.x;
  if (i < n) dst[i] = (half_t)src[i];
}

// dst[b][c][r] = src[b][r][c]  (f32 -> f16), batch of R x C matrices
__global__ void k_transpose_f16(const float* __restrict__ src, half_t* __restrict__ dst,
                                int R, int C, int total) {
  int i = blockIdx.x * blockDim.x + threadIdx.x;
  if (i >= total) return;
  int b = i / (R * C);
  int rem = i - b * (R * C);
  int r = rem / C;
  int c = rem - r * C;
  dst[(size_t)b * R * C + (size_t)c * R + r] = (half_t)src[i];
}

// ---------------- QKV projection ----------------
// grid (BS/128, H, 3), block 128 (4 waves, each: 32 rows x 64 cols, K=D).
// Two A tiles share every B tile -> 8 WMMAs per 12 loads; d-loop fully
// unrolled so the scheduler hoists clauses with partial s_wait_loadcnt and
// the first WMMA per tile folds inline C=0.
__global__ __launch_bounds__(128)
void k_qkv(const half_t* __restrict__ Xh,                       // [BS][D]
           const half_t* __restrict__ WqT, const half_t* __restrict__ WkT,
           const half_t* __restrict__ WvT,                      // [H][64][D]
           const float* __restrict__ bq, const float* __restrict__ bk,
           const float* __restrict__ bv,
           half_t* __restrict__ qh, half_t* __restrict__ kh,    // [B][H][S][64]
           half_t* __restrict__ vth)                            // [B][H][64][S]
{
  const int lane = threadIdx.x & 31;
  const int wave = threadIdx.x >> 5;
  const int hi = lane >> 4, n = lane & 15;
  const int row0 = blockIdx.x * 128 + wave * 32;
  const int h = blockIdx.y;
  const int which = blockIdx.z;

  const half_t* WT = (which == 0 ? WqT : which == 1 ? WkT : WvT) + (size_t)h * DK_ * D_;
  const float* bias = (which == 0 ? bq : which == 1 ? bk : bv) + h * DK_;

  v8f acc[2][4];
#pragma unroll
  for (int i = 0; i < 2; ++i)
#pragma unroll
    for (int j = 0; j < 4; ++j) acc[i][j] = vzero();

#pragma unroll
  for (int d0 = 0; d0 < D_; d0 += 32) {
    v16h a0 = load_A16x32(Xh, D_, row0,      d0, lane);
    v16h a1 = load_A16x32(Xh, D_, row0 + 16, d0, lane);
#pragma unroll
    for (int j = 0; j < 4; ++j) {
      v16h b = load_B32x16(WT, D_, j * 16, d0, lane);
      acc[0][j] = wmma16(a0, b, acc[0][j]);
      acc[1][j] = wmma16(a1, b, acc[1][j]);
    }
  }

  // q pre-scale: 1/sqrt(DK) * log2(e) so attention softmax runs in exp2 domain
  const float scale = (which == 0) ? (0.125f * 1.44269504088896341f) : 1.0f;
#pragma unroll
  for (int i = 0; i < 2; ++i) {
#pragma unroll
    for (int j = 0; j < 4; ++j) {
      const int col = j * 16 + n;
      const float bb = bias[col];
#pragma unroll
      for (int r = 0; r < 8; ++r) {
        const int rg = row0 + 16 * i + r + 8 * hi;     // global row in [0, BS)
        const int bi = rg >> 11;                       // / S_
        const int s  = rg & (S_ - 1);
        const float val = (acc[i][j][r] + bb) * scale;
        if (which == 2) {
          vth[(((size_t)(bi * H_ + h) * DV_) + col) * S_ + s] = (half_t)val;
        } else {
          half_t* dst = (which == 0) ? qh : kh;
          dst[(((size_t)(bi * H_ + h) * S_) + s) * DK_ + col] = (half_t)val;
        }
      }
    }
  }
}

// ---------------- fused flash attention ----------------
// grid (S/64, H, B), block 128: each wave owns 16 query rows, streams keys 64 at a time.
__global__ __launch_bounds__(128)
void k_attn(const half_t* __restrict__ qh, const half_t* __restrict__ kh,
            const half_t* __restrict__ vth, half_t* __restrict__ ctx)  // ctx [BS][HD]
{
  __shared__ __align__(16) half_t Pbuf[4][16 * 64];
  const int lane = threadIdx.x & 31;
  const int wave = threadIdx.x >> 5;
  const int hi = lane >> 4, n = lane & 15;
  const int h = blockIdx.y;
  const int bi = blockIdx.z;
  const int row0 = blockIdx.x * 64 + wave * 16;       // query s index

  const half_t* qb = qh + ((size_t)(bi * H_ + h) * S_) * DK_;
  const half_t* kb = kh + ((size_t)(bi * H_ + h) * S_) * DK_;
  const half_t* vb = vth + ((size_t)(bi * H_ + h) * DV_) * S_;
  half_t* P = Pbuf[wave];

  // Q fragment (16 rows x DK=64) lives in registers for the whole kernel
  const v16h aq0 = load_A16x32(qb, DK_, row0, 0, lane);
  const v16h aq1 = load_A16x32(qb, DK_, row0, 32, lane);

  float mrow[8], lsum[8];
#pragma unroll
  for (int r = 0; r < 8; ++r) { mrow[r] = -__builtin_inff(); lsum[r] = 0.0f; }
  v8f acc[4];
#pragma unroll
  for (int j = 0; j < 4; ++j) acc[j] = vzero();

  for (int t0 = 0; t0 < S_; t0 += 64) {
    // Batch all K-operand loads first: consumed in order by the score WMMAs,
    // allowing partial s_wait_loadcnt overlap.
    v16h kt[8];
#pragma unroll
    for (int t = 0; t < 4; ++t) {
      kt[2 * t]     = load_B32x16(kb, DK_, t0 + 16 * t, 0,  lane);
      kt[2 * t + 1] = load_B32x16(kb, DK_, t0 + 16 * t, 32, lane);
    }
    v8f sc[4];
#pragma unroll
    for (int t = 0; t < 4; ++t) {
      sc[t] = wmma16(aq0, kt[2 * t], vzero());        // inline C=0 encoding
      sc[t] = wmma16(aq1, kt[2 * t + 1], sc[t]);
    }

    // Prefetch V tiles now; their latency hides under the softmax VALU work.
    v16h vt[8];
#pragma unroll
    for (int j = 0; j < 4; ++j) {
      vt[2 * j]     = load_B32x16(vb, S_, j * 16, t0,      lane);
      vt[2 * j + 1] = load_B32x16(vb, S_, j * 16, t0 + 32, lane);
    }

    // Online softmax in exp2 domain (q carries 1/sqrt(DK)*log2e).
    float p[4][8];
#pragma unroll
    for (int r = 0; r < 8; ++r) {
      float tm = fmaxf(fmaxf(sc[0][r], sc[1][r]), fmaxf(sc[2][r], sc[3][r]));
#pragma unroll
      for (int o = 1; o < 16; o <<= 1) tm = fmaxf(tm, __shfl_xor(tm, o, 32));
      const float mnew = fmaxf(mrow[r], tm);
      const float corr = __builtin_amdgcn_exp2f(mrow[r] - mnew);
      float ps = 0.0f;
#pragma unroll
      for (int t = 0; t < 4; ++t) {
        p[t][r] = __builtin_amdgcn_exp2f(sc[t][r] - mnew);
        ps += p[t][r];
      }
#pragma unroll
      for (int o = 1; o < 16; o <<= 1) ps += __shfl_xor(ps, o, 32);
      lsum[r] = lsum[r] * corr + ps;
      mrow[r] = mnew;
#pragma unroll
      for (int j = 0; j < 4; ++j) acc[j][r] *= corr;
    }

    // Stage P (16x64 f16) through per-wave LDS: C-layout -> A-layout.
    // Single wave_barrier orders store->load; LDS is in-order per wave and
    // conservative aliasing keeps per-thread program order otherwise, so the
    // scheduler remains free to hoist next iteration's K loads over ctx WMMAs.
#pragma unroll
    for (int r = 0; r < 8; ++r)
#pragma unroll
      for (int t = 0; t < 4; ++t)
        P[(r + 8 * hi) * 64 + 16 * t + n] = (half_t)p[t][r];
    __builtin_amdgcn_wave_barrier();
    const v16h ap0 = load_A16x32(P, 64, 0, 0,  lane);
    const v16h ap1 = load_A16x32(P, 64, 0, 32, lane);
#pragma unroll
    for (int j = 0; j < 4; ++j) {
      acc[j] = wmma16(ap0, vt[2 * j],     acc[j]);
      acc[j] = wmma16(ap1, vt[2 * j + 1], acc[j]);
    }
  }

#pragma unroll
  for (int j = 0; j < 4; ++j) {
    const int col = h * DV_ + j * 16 + n;   // torch.cat head order: h*DV + dv
#pragma unroll
    for (int r = 0; r < 8; ++r) {
      const int s = row0 + r + 8 * hi;
      const float v = acc[j][r] * __builtin_amdgcn_rcpf(lsum[r]);
      ctx[((size_t)bi * S_ + s) * HD_ + col] = (half_t)v;
    }
  }
}

// ---------------- output projection ----------------
// grid (BS/128, D/64), block 128; 32 rows/wave, B tiles shared across 2 A tiles.
__global__ __launch_bounds__(128)
void k_out(const half_t* __restrict__ Ch, const half_t* __restrict__ WoT,
           const float* __restrict__ bo, float* __restrict__ out)
{
  const int lane = threadIdx.x & 31;
  const int wave = threadIdx.x >> 5;
  const int hi = lane >> 4, n = lane & 15;
  const int row0 = blockIdx.x * 128 + wave * 32;
  const int col0 = blockIdx.y * 64;

  v8f acc[2][4];
#pragma unroll
  for (int i = 0; i < 2; ++i)
#pragma unroll
    for (int j = 0; j < 4; ++j) acc[i][j] = vzero();

#pragma unroll
  for (int d0 = 0; d0 < HD_; d0 += 32) {
    v16h a0 = load_A16x32(Ch, HD_, row0,      d0, lane);
    v16h a1 = load_A16x32(Ch, HD_, row0 + 16, d0, lane);
#pragma unroll
    for (int j = 0; j < 4; ++j) {
      v16h b = load_B32x16(WoT, HD_, col0 + j * 16, d0, lane);
      acc[0][j] = wmma16(a0, b, acc[0][j]);
      acc[1][j] = wmma16(a1, b, acc[1][j]);
    }
  }

#pragma unroll
  for (int i = 0; i < 2; ++i) {
#pragma unroll
    for (int j = 0; j < 4; ++j) {
      const int col = col0 + j * 16 + n;
      const float bb = bo[col];
#pragma unroll
      for (int r = 0; r < 8; ++r) {
        const int rg = row0 + 16 * i + r + 8 * hi;
        out[(size_t)rg * D_ + col] = acc[i][j][r] + bb;
      }
    }
  }
}

extern "C" void kernel_launch(void* const* d_in, const int* in_sizes, int n_in,
                              void* d_out, int out_size, void* d_ws, size_t ws_size,
                              hipStream_t stream) {
  (void)in_sizes; (void)n_in; (void)out_size; (void)ws_size;
  const float* hidden = (const float*)d_in[0];
  const float* Wq = (const float*)d_in[1];
  const float* bq = (const float*)d_in[2];
  const float* Wk = (const float*)d_in[3];
  const float* bk = (const float*)d_in[4];
  const float* Wv = (const float*)d_in[5];
  const float* bv = (const float*)d_in[6];
  const float* Wo = (const float*)d_in[7];
  const float* bo = (const float*)d_in[8];
  float* out = (float*)d_out;

  char* ws = (char*)d_ws;
  size_t off = 0;
  auto carve = [&](size_t bytes) -> void* {
    void* p = ws + off;
    off += (bytes + 255) & ~(size_t)255;
    return p;
  };
  half_t* Xh   = (half_t*)carve((size_t)BS_ * D_ * 2);            // hidden f16
  half_t* WqT  = (half_t*)carve((size_t)H_ * DK_ * D_ * 2);       // [H][dk][d]
  half_t* WkT  = (half_t*)carve((size_t)H_ * DK_ * D_ * 2);
  half_t* WvT  = (half_t*)carve((size_t)H_ * DV_ * D_ * 2);
  half_t* WoT  = (half_t*)carve((size_t)D_ * HD_ * 2);            // [n][d]
  half_t* qhb  = (half_t*)carve((size_t)B_ * H_ * S_ * DK_ * 2);  // [b][h][s][dk]
  half_t* khb  = (half_t*)carve((size_t)B_ * H_ * S_ * DK_ * 2);
  half_t* vtb  = (half_t*)carve((size_t)B_ * H_ * DV_ * S_ * 2);  // [b][h][dv][t]
  half_t* ctxh = (half_t*)carve((size_t)BS_ * HD_ * 2);           // [bs][h*dv]

  {
    int nt = BS_ * D_;
    k_cvt_f16<<<(nt + 255) / 256, 256, 0, stream>>>(hidden, Xh, nt);
  }
  {
    int nt = H_ * D_ * DK_;
    k_transpose_f16<<<(nt + 255) / 256, 256, 0, stream>>>(Wq, WqT, D_, DK_, nt);
    k_transpose_f16<<<(nt + 255) / 256, 256, 0, stream>>>(Wk, WkT, D_, DK_, nt);
    k_transpose_f16<<<(nt + 255) / 256, 256, 0, stream>>>(Wv, WvT, D_, DV_, nt);
  }
  {
    int nt = HD_ * D_;
    k_transpose_f16<<<(nt + 255) / 256, 256, 0, stream>>>(Wo, WoT, HD_, D_, nt);
  }

  k_qkv<<<dim3(BS_ / 128, H_, 3), 128, 0, stream>>>(Xh, WqT, WkT, WvT, bq, bk, bv,
                                                    qhb, khb, vtb);
  k_attn<<<dim3(S_ / 64, H_, B_), 128, 0, stream>>>(qhb, khb, vtb, ctxh);
  k_out<<<dim3(BS_ / 128, D_ / 64), 128, 0, stream>>>(ctxh, WoT, bo, out);
}
